// VSSBlock_localmamba_42812234006794
// MI455X (gfx1250) — compile-verified, matrix operand order checked by reading, not run on hardware
//
#include <hip/hip_runtime.h>
#include <hip/hip_bf16.h>

typedef __attribute__((ext_vector_type(16))) __bf16 v16bf;
typedef __attribute__((ext_vector_type(8)))  float  v8f;

// ---------------- constants ----------------
#define CC   96
#define DI   192
#define NN   16
#define RR   6
#define KK   4
#define HH   64
#define WW   64
#define HW   4096
#define BB   8
#define MROW 32768        // B*H*W
#define XDBL_LD 48        // padded (R + 2N = 38 -> 48)

// workspace layout (bytes)
static constexpr size_t O_XN   = 0;                     // bf16 32768*96  (reused: YLN bf16 32768*192)
static constexpr size_t O_XC   = 12582912;              // f32 (B,192,HW) (reused: YACC (B,HW,192))
static constexpr size_t O_Z    = O_XC   + 25165824;     // f32 (M,192)
static constexpr size_t O_XCT  = O_Z    + 25165824;     // f32 (B,HW,192)
static constexpr size_t O_XDBL = O_XCT  + 25165824;     // f32 (K, M, 48)
static constexpr size_t O_XRES = O_XDBL + 25165824;     // f32 (M,96)
static constexpr size_t O_XL   = O_XRES + 12582912;     // f32 (M,96)
static constexpr size_t O_H1   = O_XL   + 12582912;     // f32 (B,32,HW)
static constexpr size_t O_H2   = O_H1   + 4194304;      // f32 (B,96,HW)
static constexpr size_t O_P    = O_H2   + 12582912;     // f32 (B,96)
static constexpr size_t O_A    = O_P    + 4096;         // f32 (B,96)

// ---------------- helpers ----------------
__device__ __forceinline__ v8f wmma_bf16(v16bf a, v16bf b, v8f c) {
  return __builtin_amdgcn_wmma_f32_16x16x32_bf16(false, a, false, b, (short)0, c, false, false);
}

__device__ __forceinline__ float wave_sum32(float v) {
#pragma unroll
  for (int off = 16; off >= 1; off >>= 1) v += __shfl_xor(v, off, 32);
  return v;
}

// spatial position for scan-direction k at sequence index l (all dims pow2)
__device__ __forceinline__ int posmap(int k, int l) {
  if (k & 2) l = 4095 - l;
  int h, w;
  if ((k & 1) == 0) {
    int hg = (l >> 9) & 7, wg = (l >> 6) & 7, hi = (l >> 3) & 7, wi = l & 7;
    h = hg * 8 + hi; w = wg * 8 + wi;
  } else {
    int wg = (l >> 9) & 7, hg = (l >> 6) & 7, wi = (l >> 3) & 7, hi = l & 7;
    h = hg * 8 + hi; w = wg * 8 + wi;
  }
  return h * 64 + w;
}

__device__ __forceinline__ float silu(float x) { return x / (1.f + __expf(-x)); }

// ---------------- K1: LayerNorm-1 -> bf16 (M,96). wave per row ----------------
__global__ void k_ln1(const float* __restrict__ in, const float* __restrict__ w,
                      const float* __restrict__ bia, __bf16* __restrict__ xn) {
  int lane = threadIdx.x & 31;
  int row  = blockIdx.x * 8 + (threadIdx.x >> 5);
  int b = row >> 12, pos = row & 4095;
  float v[3]; float s = 0.f, sq = 0.f;
#pragma unroll
  for (int i = 0; i < 3; ++i) {
    int c = lane + 32 * i;
    v[i] = in[(b * CC + c) * HW + pos];
    s += v[i]; sq += v[i] * v[i];
  }
  s = wave_sum32(s); sq = wave_sum32(sq);
  float mu = s * (1.f / CC);
  float var = sq * (1.f / CC) - mu * mu;
  float rs = rsqrtf(var + 1e-5f);
#pragma unroll
  for (int i = 0; i < 3; ++i) {
    int c = lane + 32 * i;
    xn[row * CC + c] = (__bf16)((v[i] - mu) * rs * w[c] + bia[c]);
  }
}

// ---------------- K2: in_proj GEMM (M=32768,N=384,K=96) WMMA bf16 ----------------
__global__ void k_inproj(const __bf16* __restrict__ xn, const float* __restrict__ W,
                         float* __restrict__ xc, float* __restrict__ z) {
  int lane = threadIdx.x & 31;
  int wv = threadIdx.x >> 5;
  int wm = wv & 1, wn = wv >> 1;                 // 2 x 4 waves
  int tileM = blockIdx.x * 32 + wm * 16;
  int tileN = blockIdx.y * 64 + wn * 16;
  int row = tileM + (lane & 15);
  int col = tileN + (lane & 15);
  int khalf = (lane & 16) ? 8 : 0;
  v8f acc = {};
  for (int k0 = 0; k0 < CC; k0 += 32) {
    if (k0 + 32 < CC) __builtin_prefetch(&xn[row * CC + k0 + 32], 0, 1);
    v16bf a, bb;
#pragma unroll
    for (int j = 0; j < 16; ++j) {
      int kk = k0 + (j & 7) + ((j >> 3) << 4) + khalf;
      a[j]  = xn[row * CC + kk];
      bb[j] = (__bf16)W[col * CC + kk];
    }
    acc = wmma_bf16(a, bb, acc);
  }
  int mbase = tileM + ((lane & 16) ? 8 : 0);
#pragma unroll
  for (int r = 0; r < 8; ++r) {
    int m = mbase + r, e = col;
    float vv = acc[r];
    int b = m >> 12, pos = m & 4095;
    if (e < DI) xc[(b * DI + e) * HW + pos] = vv;            // (B,DI,HW) for conv
    else        z[m * DI + (e - DI)] = vv;                   // (M,DI)
  }
}

// ---------------- K3: depthwise 3x3 + bias + SiLU -> transposed (B,HW,DI) ----------------
__global__ void k_dwconv(const float* __restrict__ xc, const float* __restrict__ cw,
                         const float* __restrict__ cb, float* __restrict__ xct) {
  int idx = blockIdx.x * blockDim.x + threadIdx.x;      // ((b*DI+d)<<12)+pos
  int pos = idx & 4095;
  int bd  = idx >> 12;
  int d = bd % DI, b = bd / DI;
  int h = pos >> 6, w = pos & 63;
  const float* src = xc + (size_t)bd * HW;
  float acc = cb[d];
#pragma unroll
  for (int kh = 0; kh < 3; ++kh) {
    int hh = h + kh - 1;
    if ((unsigned)hh >= 64u) continue;
#pragma unroll
    for (int kw = 0; kw < 3; ++kw) {
      int ww2 = w + kw - 1;
      if ((unsigned)ww2 >= 64u) continue;
      acc += src[hh * 64 + ww2] * cw[d * 9 + kh * 3 + kw];
    }
  }
  acc = silu(acc);
  xct[((size_t)(b << 12) + pos) * DI + d] = acc;
}

// ---------------- K4: x_proj GEMM per direction (M=32768,N=48pad,K=192) ----------------
__global__ void k_xproj(const float* __restrict__ xct, const float* __restrict__ xw,
                        float* __restrict__ xdbl) {
  int kdir = blockIdx.z;
  int lane = threadIdx.x & 31;
  int wv = threadIdx.x >> 5;                     // 0..2 -> N tile
  int tileM = blockIdx.x * 16;
  int cbase = wv * 16;
  int rlane = lane & 15;
  int khalf = (lane & 16) ? 8 : 0;
  int m = tileM + rlane;
  int b = m >> 12, l = m & 4095;
  const float* Arow = xct + (size_t)((b << 12) + posmap(kdir, l)) * DI;
  int c = cbase + rlane;
  const float* Wk = xw + kdir * (RR + 2 * NN) * DI;
  bool cvalid = c < (RR + 2 * NN);
  v8f acc = {};
  for (int k0 = 0; k0 < DI; k0 += 32) {
    if (k0 + 32 < DI) __builtin_prefetch(&Arow[k0 + 32], 0, 1);
    v16bf a, bb;
#pragma unroll
    for (int j = 0; j < 16; ++j) {
      int kk = k0 + (j & 7) + ((j >> 3) << 4) + khalf;
      a[j]  = (__bf16)Arow[kk];
      bb[j] = cvalid ? (__bf16)Wk[c * DI + kk] : (__bf16)0.0f;
    }
    acc = wmma_bf16(a, bb, acc);
  }
  int mbase = tileM + ((lane & 16) ? 8 : 0);
#pragma unroll
  for (int r = 0; r < 8; ++r) {
    int mr = mbase + r;
    xdbl[(size_t)((kdir << 15) + mr) * XDBL_LD + c] = acc[r];
  }
}

// ---------------- K5: selective scan. lane=(n), 16-lane groups per d ----------------
__global__ void k_scan(const float* __restrict__ xdbl, const float* __restrict__ xct,
                       const float* __restrict__ dtw, const float* __restrict__ dtb,
                       const float* __restrict__ alog, const float* __restrict__ Dsp,
                       float* __restrict__ yacc) {
  int lane = threadIdx.x & 31;
  int wid  = blockIdx.x * (blockDim.x >> 5) + (threadIdx.x >> 5);  // 0..3071
  int dpair = wid % 96;
  int bk    = wid / 96;
  int k = bk & 3, b = bk >> 2;
  int n = lane & 15;
  int d = dpair * 2 + (lane >> 4);
  int kd = k * DI + d;
  float An = -__expf(alog[kd * NN + n]);
  float wr[RR];
#pragma unroll
  for (int r = 0; r < RR; ++r) wr[r] = dtw[kd * RR + r];
  float bias = dtb[kd];
  float Dd   = Dsp[kd];
  float h = 0.f;
  for (int l = 0; l < 4096; ++l) {
    const float* rowp = xdbl + (size_t)((k << 15) | (b << 12) | l) * XDBL_LD;
    float dtr = bias;
#pragma unroll
    for (int r = 0; r < RR; ++r) dtr += rowp[r] * wr[r];
    float dt = (dtr > 20.f) ? dtr : log1pf(__expf(dtr));   // softplus
    int pos = posmap(k, l);
    float u  = xct[((size_t)(b << 12) + pos) * DI + d];
    float Bn = rowp[RR + n];
    float Cn = rowp[RR + NN + n];
    h = __expf(dt * An) * h + (dt * u) * Bn;
    float yc = h * Cn;
    yc += __shfl_xor(yc, 1, 16);
    yc += __shfl_xor(yc, 2, 16);
    yc += __shfl_xor(yc, 4, 16);
    yc += __shfl_xor(yc, 8, 16);
    if (n == 0) atomicAdd(&yacc[((size_t)(b << 12) + pos) * DI + d], yc + Dd * u);
  }
}

// ---------------- K6: out_norm LN(192) * silu(z) -> bf16 (M,192) ----------------
__global__ void k_outnorm(const float* __restrict__ y, const float* __restrict__ z,
                          const float* __restrict__ w, const float* __restrict__ bia,
                          __bf16* __restrict__ yln) {
  int lane = threadIdx.x & 31;
  int row  = blockIdx.x * 8 + (threadIdx.x >> 5);
  const float* yr = y + (size_t)row * DI;
  float v[6]; float s = 0.f, sq = 0.f;
#pragma unroll
  for (int i = 0; i < 6; ++i) {
    int c = lane + 32 * i;
    v[i] = yr[c];
    s += v[i]; sq += v[i] * v[i];
  }
  s = wave_sum32(s); sq = wave_sum32(sq);
  float mu = s * (1.f / DI);
  float var = sq * (1.f / DI) - mu * mu;
  float rs = rsqrtf(var + 1e-5f);
#pragma unroll
  for (int i = 0; i < 6; ++i) {
    int c = lane + 32 * i;
    float g = silu(z[(size_t)row * DI + c]);
    yln[(size_t)row * DI + c] = (__bf16)(((v[i] - mu) * rs * w[c] + bia[c]) * g);
  }
}

// ---------------- K7: out_proj GEMM (M=32768,N=96,K=192) + skip ----------------
__global__ void k_outproj(const __bf16* __restrict__ yln, const float* __restrict__ W,
                          const float* __restrict__ in, const float* __restrict__ sk,
                          float* __restrict__ xres) {
  int lane = threadIdx.x & 31;
  int wv = threadIdx.x >> 5;                     // 0..11
  int wm = wv & 1, wn = wv >> 1;                 // 2 x 6 waves
  int tileM = blockIdx.x * 32 + wm * 16;
  int tileN = wn * 16;
  int row = tileM + (lane & 15);
  int col = tileN + (lane & 15);
  int khalf = (lane & 16) ? 8 : 0;
  v8f acc = {};
  for (int k0 = 0; k0 < DI; k0 += 32) {
    if (k0 + 32 < DI) __builtin_prefetch(&yln[(size_t)row * DI + k0 + 32], 0, 1);
    v16bf a, bb;
#pragma unroll
    for (int j = 0; j < 16; ++j) {
      int kk = k0 + (j & 7) + ((j >> 3) << 4) + khalf;
      a[j]  = yln[(size_t)row * DI + kk];
      bb[j] = (__bf16)W[col * DI + kk];
    }
    acc = wmma_bf16(a, bb, acc);
  }
  int mbase = tileM + ((lane & 16) ? 8 : 0);
#pragma unroll
  for (int r = 0; r < 8; ++r) {
    int m = mbase + r, c = col;
    int b = m >> 12, pos = m & 4095;
    float shortcut = in[(b * CC + c) * HW + pos];
    xres[(size_t)m * CC + c] = shortcut * sk[c] + acc[r];
  }
}

// ---------------- K8: LayerNorm-2 -> f32 (M,96) ----------------
__global__ void k_ln2(const float* __restrict__ x, const float* __restrict__ w,
                      const float* __restrict__ bia, float* __restrict__ xl) {
  int lane = threadIdx.x & 31;
  int row  = blockIdx.x * 8 + (threadIdx.x >> 5);
  const float* xr = x + (size_t)row * CC;
  float v[3]; float s = 0.f, sq = 0.f;
#pragma unroll
  for (int i = 0; i < 3; ++i) {
    int c = lane + 32 * i;
    v[i] = xr[c];
    s += v[i]; sq += v[i] * v[i];
  }
  s = wave_sum32(s); sq = wave_sum32(sq);
  float mu = s * (1.f / CC);
  float var = sq * (1.f / CC) - mu * mu;
  float rs = rsqrtf(var + 1e-5f);
#pragma unroll
  for (int i = 0; i < 3; ++i) {
    int c = lane + 32 * i;
    xl[(size_t)row * CC + c] = (v[i] - mu) * rs * w[c] + bia[c];
  }
}

// ---------------- K9: CAB conv1 (96->32, 3x3) + GELU ----------------
__global__ void k_cab1(const float* __restrict__ xl, const float* __restrict__ cw,
                       const float* __restrict__ cb, float* __restrict__ h1) {
  int idx = blockIdx.x * blockDim.x + threadIdx.x;   // ((b*32+o)<<12)+pos
  int pos = idx & 4095;
  int bo  = idx >> 12;
  int o = bo & 31, b = bo >> 5;
  int h = pos >> 6, w = pos & 63;
  float acc = cb[o];
#pragma unroll
  for (int kh = 0; kh < 3; ++kh) {
    int hh = h + kh - 1;
    if ((unsigned)hh >= 64u) continue;
#pragma unroll
    for (int kw = 0; kw < 3; ++kw) {
      int ww2 = w + kw - 1;
      if ((unsigned)ww2 >= 64u) continue;
      const float* row = xl + ((size_t)(b << 12) + hh * 64 + ww2) * CC;
      const float* wt  = cw + (size_t)o * CC * 9 + kh * 3 + kw;
      for (int ci = 0; ci < CC; ++ci) acc += row[ci] * wt[(size_t)ci * 9];
    }
  }
  float g = 0.5f * acc * (1.f + erff(acc * 0.70710678118654752f));
  h1[(size_t)bo * HW + pos] = g;
}

// ---------------- K10: CAB conv2 (32->96, 3x3) + bias ----------------
__global__ void k_cab2(const float* __restrict__ h1, const float* __restrict__ cw,
                       const float* __restrict__ cb, float* __restrict__ h2) {
  int idx = blockIdx.x * blockDim.x + threadIdx.x;   // ((b*96+c)<<12)+pos
  int pos = idx & 4095;
  int bc  = idx >> 12;
  int c = bc % CC, b = bc / CC;
  int h = pos >> 6, w = pos & 63;
  float acc = cb[c];
#pragma unroll
  for (int kh = 0; kh < 3; ++kh) {
    int hh = h + kh - 1;
    if ((unsigned)hh >= 64u) continue;
#pragma unroll
    for (int kw = 0; kw < 3; ++kw) {
      int ww2 = w + kw - 1;
      if ((unsigned)ww2 >= 64u) continue;
      for (int ci = 0; ci < 32; ++ci)
        acc += h1[((size_t)(b * 32 + ci) << 12) + hh * 64 + ww2] *
               cw[((size_t)c * 32 + ci) * 9 + kh * 3 + kw];
    }
  }
  h2[(size_t)bc * HW + pos] = acc;
}

// ---------------- K11a: global average pool per (b,c) ----------------
__global__ void k_pool(const float* __restrict__ h2, float* __restrict__ p) {
  int bid = blockIdx.x;                      // b*96+c
  const float* src = h2 + (size_t)bid * HW;
  float s = 0.f;
  for (int i = threadIdx.x; i < HW; i += 256) s += src[i];
  s = wave_sum32(s);
  __shared__ float sm[8];
  int wv = threadIdx.x >> 5, lane = threadIdx.x & 31;
  if (lane == 0) sm[wv] = s;
  __syncthreads();
  if (threadIdx.x == 0) {
    float t = 0.f;
    for (int i = 0; i < 8; ++i) t += sm[i];
    p[bid] = t * (1.f / HW);
  }
}

// ---------------- K11b: channel attention MLP ----------------
__global__ void k_ca(const float* __restrict__ p, const float* __restrict__ w1,
                     const float* __restrict__ b1, const float* __restrict__ w2,
                     const float* __restrict__ b2, float* __restrict__ a) {
  int b = blockIdx.x, c = threadIdx.x;       // 96 threads
  __shared__ float a1s[3];
  if (c < 3) {
    float s = b1[c];
    for (int i = 0; i < CC; ++i) s += p[b * CC + i] * w1[c * CC + i];
    a1s[c] = fmaxf(s, 0.f);
  }
  __syncthreads();
  float s = b2[c];
#pragma unroll
  for (int j = 0; j < 3; ++j) s += a1s[j] * w2[c * 3 + j];
  a[b * CC + c] = 1.f / (1.f + __expf(-s));
}

// ---------------- K12: final combine -> (B,C,H,W) ----------------
__global__ void k_final(const float* __restrict__ xres, const float* __restrict__ h2,
                        const float* __restrict__ a, const float* __restrict__ sk2,
                        float* __restrict__ out) {
  int idx = blockIdx.x * blockDim.x + threadIdx.x;   // ((b*96+c)<<12)+pos
  int pos = idx & 4095;
  int bc  = idx >> 12;
  int c = bc % CC, b = bc / CC;
  out[idx] = xres[((size_t)(b << 12) + pos) * CC + c] * sk2[c] +
             h2[idx] * a[b * CC + c];
}

// ---------------- host launcher ----------------
extern "C" void kernel_launch(void* const* d_in, const int* in_sizes, int n_in,
                              void* d_out, int out_size, void* d_ws, size_t ws_size,
                              hipStream_t stream) {
  (void)in_sizes; (void)n_in; (void)out_size; (void)ws_size;
  const float* input    = (const float*)d_in[0];
  const float* ln1_w    = (const float*)d_in[1];
  const float* ln1_b    = (const float*)d_in[2];
  const float* inproj_w = (const float*)d_in[3];
  const float* conv_w   = (const float*)d_in[4];
  const float* conv_b   = (const float*)d_in[5];
  const float* xproj_w  = (const float*)d_in[6];
  const float* dtp_w    = (const float*)d_in[7];
  const float* dtp_b    = (const float*)d_in[8];
  const float* A_logs   = (const float*)d_in[9];
  const float* Ds       = (const float*)d_in[10];
  const float* onw      = (const float*)d_in[11];
  const float* onb      = (const float*)d_in[12];
  const float* outproj_w= (const float*)d_in[13];
  const float* skip1    = (const float*)d_in[14];
  const float* skip2    = (const float*)d_in[15];
  const float* ln2_w    = (const float*)d_in[16];
  const float* ln2_b    = (const float*)d_in[17];
  const float* cab1_w   = (const float*)d_in[18];
  const float* cab1_b   = (const float*)d_in[19];
  const float* cab2_w   = (const float*)d_in[20];
  const float* cab2_b   = (const float*)d_in[21];
  const float* ca1_w    = (const float*)d_in[22];
  const float* ca1_b    = (const float*)d_in[23];
  const float* ca2_w    = (const float*)d_in[24];
  const float* ca2_b    = (const float*)d_in[25];
  float* out = (float*)d_out;

  char* ws = (char*)d_ws;
  __bf16* xn   = (__bf16*)(ws + O_XN);
  float*  xc   = (float*)(ws + O_XC);
  float*  yacc = (float*)(ws + O_XC);   // reuse after dwconv
  float*  z    = (float*)(ws + O_Z);
  float*  xct  = (float*)(ws + O_XCT);
  float*  xdbl = (float*)(ws + O_XDBL);
  __bf16* yln  = (__bf16*)(ws + O_XN);  // reuse after in_proj
  float*  xres = (float*)(ws + O_XRES);
  float*  xl   = (float*)(ws + O_XL);
  float*  h1   = (float*)(ws + O_H1);
  float*  h2   = (float*)(ws + O_H2);
  float*  p    = (float*)(ws + O_P);
  float*  a    = (float*)(ws + O_A);

  // 1) LN1
  k_ln1<<<MROW / 8, 256, 0, stream>>>(input, ln1_w, ln1_b, xn);
  // 2) in_proj GEMM (WMMA)
  k_inproj<<<dim3(MROW / 32, 384 / 64), 256, 0, stream>>>(xn, inproj_w, xc, z);
  // 3) depthwise conv + SiLU
  k_dwconv<<<(BB * DI * HW) / 256, 256, 0, stream>>>(xc, conv_w, conv_b, xct);
  // zero the y accumulator (reuses xc region, now free)
  hipMemsetAsync(yacc, 0, (size_t)MROW * DI * sizeof(float), stream);
  // 4) x_proj GEMM per direction (WMMA)
  k_xproj<<<dim3(MROW / 16, 1, KK), 96, 0, stream>>>(xct, xproj_w, xdbl);
  // 5) selective scan (lane-parallel over (b,k,d,n); atomic combine over k)
  k_scan<<<384, 256, 0, stream>>>(xdbl, xct, dtp_w, dtp_b, A_logs, Ds, yacc);
  // 6) out_norm * silu(z)
  k_outnorm<<<MROW / 8, 256, 0, stream>>>(yacc, z, onw, onb, yln);
  // 7) out_proj GEMM + residual (WMMA)
  k_outproj<<<MROW / 32, 384, 0, stream>>>(yln, outproj_w, input, skip1, xres);
  // 8) LN2
  k_ln2<<<MROW / 8, 256, 0, stream>>>(xres, ln2_w, ln2_b, xl);
  // 9) CAB conv1 + GELU
  k_cab1<<<(BB * 32 * HW) / 256, 256, 0, stream>>>(xl, cab1_w, cab1_b, h1);
  // 10) CAB conv2
  k_cab2<<<(BB * CC * HW) / 256, 256, 0, stream>>>(h1, cab2_w, cab2_b, h2);
  // 11) channel attention
  k_pool<<<BB * CC, 256, 0, stream>>>(h2, p);
  k_ca<<<BB, 96, 0, stream>>>(p, ca1_w, ca1_b, ca2_w, ca2_b, a);
  // 12) final combine
  k_final<<<(BB * CC * HW) / 256, 256, 0, stream>>>(xres, h2, a, skip2, out);
}
